// MultiHeadedAttention_7516192768864
// MI455X (gfx1250) — matrix operand order repaired and checked by reading, not compiled, not run on hardware
//
#include <hip/hip_runtime.h>

#define S_LEN  2048
#define DMODEL 512
#define HEADS  8
#define DK     64
#define BATCH  2
#define EPS_MASK 1000000000.0f
#define NCHUNK (S_LEN / 32)
#define NKB    (DMODEL / 32)

#if __has_builtin(__builtin_amdgcn_global_load_async_to_lds_b128) && \
    __has_builtin(__builtin_amdgcn_s_wait_asynccnt)
#define ASYNC_LDS 1
#else
#define ASYNC_LDS 0
#endif

typedef _Float16 half_t;
typedef __attribute__((ext_vector_type(16))) _Float16 v16h;
typedef __attribute__((ext_vector_type(8)))  _Float16 v8h;
typedef __attribute__((ext_vector_type(4)))  _Float16 v4h;
typedef __attribute__((ext_vector_type(2)))  _Float16 v2h;
typedef __attribute__((ext_vector_type(8)))  float    v8f;
typedef __attribute__((ext_vector_type(4)))  float    v4f;
typedef __attribute__((ext_vector_type(4)))  int      v4i;

// async-to-LDS builtin takes int4 pointers: global = addrspace(1), LDS = addrspace(3)
#define AS1I4(p) ((__attribute__((address_space(1))) v4i*)(p))
#define AS3I4(p) ((__attribute__((address_space(3))) v4i*)(p))

// ---------------- WMMA fragment helpers (wave32, 16x16x32 f16) ----------------
// A (16xK): lane m = lane%16, element e -> k = e + 16*(lane/16)
__device__ __forceinline__ v16h load_frag_a(const half_t* base, int row0, int pitch, int koff) {
  const int lane = threadIdx.x & 31;
  const int m = lane & 15, hi = lane >> 4;
  const half_t* r = base + (row0 + m) * pitch + koff;
  v16h a;
#pragma unroll
  for (int ep = 0; ep < 8; ++ep) {
    const int e = ep * 2;
    const int c = e + (hi << 4);
    v2h p = *(const v2h*)(r + c);
    a[e] = p.x; a[e + 1] = p.y;
  }
  return a;
}

// B (Kx16): lane n = lane%16, element e -> k = e + 16*(lane/16); layout base[(n0+n)*pitch + k]
__device__ __forceinline__ v16h load_frag_b(const half_t* base, int n0, int pitch, int koff) {
  const int lane = threadIdx.x & 31;
  const int n = lane & 15, hi = lane >> 4;
  const half_t* r = base + (n0 + n) * pitch + koff + (hi << 4);
  v16h b;
#pragma unroll
  for (int ep = 0; ep < 8; ++ep) {
    v2h p = *(const v2h*)(r + ep * 2);
    b[2 * ep] = p.x; b[2 * ep + 1] = p.y;
  }
  return b;
}

__device__ __forceinline__ v8f wmma_f16(v16h a, v16h b, v8f c) {
  return __builtin_amdgcn_wmma_f32_16x16x32_f16(false, a, false, b, (short)0, c, false, false);
}

__device__ __forceinline__ v4h cvt4(v4f x) {
  v4h h; h.x = (half_t)x.x; h.y = (half_t)x.y; h.z = (half_t)x.z; h.w = (half_t)x.w;
  return h;
}

// ---------------- Kernel 1: fused Q/K/V projection (z = 0/1/2), double-buffered ----------------
__global__ void __launch_bounds__(128)
qkv_proj_kernel(const float* __restrict__ query, const float* __restrict__ key_,
                const float* __restrict__ value,
                const float* __restrict__ Wq, const float* __restrict__ bq,
                const float* __restrict__ Wk, const float* __restrict__ bk,
                const float* __restrict__ Wv, const float* __restrict__ bv,
                half_t* __restrict__ qh, half_t* __restrict__ kh, half_t* __restrict__ vh)
{
  __shared__ half_t sX[2][64 * 40];   // 64 rows x 32 K, pitch 40 halves
  __shared__ half_t sW[2][64 * 40];   // transposed: 64 n x 32 K

  const float* X; const float* W; const float* bias; half_t* outp;
  if (blockIdx.z == 0)      { X = query; W = Wq; bias = bq; outp = qh; }
  else if (blockIdx.z == 1) { X = key_;  W = Wk; bias = bk; outp = kh; }
  else                      { X = value; W = Wv; bias = bv; outp = vh; }

  const int h  = blockIdx.y;
  const int m0 = blockIdx.x * 64;
  const int tid = threadIdx.x, lane = tid & 31, w = tid >> 5;
  const int nl = lane & 15, hi = lane >> 4;

  v4f xr[4], wr[4];
  auto issue = [&](int kb) {
#pragma unroll
    for (int ii = 0; ii < 4; ++ii) {
      const int i = tid + ii * 128;
      const int row = i >> 3, c4 = (i & 7) * 4;
      xr[ii] = *(const v4f*)(X + (size_t)(m0 + row) * DMODEL + kb + c4);
      const int d = i >> 4, n4 = (i & 15) * 4;
      wr[ii] = *(const v4f*)(W + ((size_t)h * DMODEL + kb + d) * DK + n4);
    }
  };
  auto commit = [&](int buf) {
#pragma unroll
    for (int ii = 0; ii < 4; ++ii) {
      const int i = tid + ii * 128;
      const int row = i >> 3, c4 = (i & 7) * 4;
      *(v4h*)(&sX[buf][row * 40 + c4]) = cvt4(xr[ii]);
      const int d = i >> 4, n4 = (i & 15) * 4;
      sW[buf][(n4 + 0) * 40 + d] = (half_t)wr[ii].x;
      sW[buf][(n4 + 1) * 40 + d] = (half_t)wr[ii].y;
      sW[buf][(n4 + 2) * 40 + d] = (half_t)wr[ii].z;
      sW[buf][(n4 + 3) * 40 + d] = (half_t)wr[ii].w;
    }
  };

  v8f acc[4] = {};
  issue(0); commit(0);
  __syncthreads();
  for (int kt = 0; kt < NKB; ++kt) {
    const int cur = kt & 1;
    if (kt + 1 < NKB) issue((kt + 1) * 32);          // prefetch next chunk
    v16h a = load_frag_a(sX[cur], w * 16, 40, 0);
#pragma unroll
    for (int vt = 0; vt < 4; ++vt) {
      v16h bf = load_frag_b(sW[cur], vt * 16, 40, 0);
      acc[vt] = wmma_f16(a, bf, acc[vt]);
    }
    if (kt + 1 < NKB) commit(cur ^ 1);               // waits loadcnt after WMMAs issued
    __syncthreads();
  }
#pragma unroll
  for (int vt = 0; vt < 4; ++vt) {
    const int n = vt * 16 + nl;
    const float bb_ = bias[h * DK + n];
#pragma unroll
    for (int r = 0; r < 8; ++r) {
      const int gm = m0 + w * 16 + r + 8 * hi;
      const int bb = gm >> 11, s = gm & (S_LEN - 1);
      outp[(((size_t)bb * HEADS + h) * S_LEN + s) * DK + n] = (half_t)(acc[vt][r] + bb_);
    }
  }
}

// ---------------- Kernel 2: flash attention, double-buffered, async K/mask staging ----------------
__global__ void __launch_bounds__(128)
attention_kernel(const half_t* __restrict__ qh, const half_t* __restrict__ kh,
                 const half_t* __restrict__ vh, const float* __restrict__ mask,
                 half_t* __restrict__ ctx)
{
  __shared__ half_t sQ[64 * 72];         // 64 x 64 f16
  __shared__ half_t sK[2][32 * 72];      // K chunk [t][dk]
  __shared__ half_t sVt[2][64 * 40];     // V chunk transposed [dk][t]
  __shared__ half_t sP[4][16 * 40];      // per-wave P staging
  __shared__ float  sM[2][64 * 36];      // mask tile [query][t]

  const int b = blockIdx.z, h = blockIdx.y, s0 = blockIdx.x * 64;
  const int tid = threadIdx.x, lane = tid & 31, w = tid >> 5;
  const int nl = lane & 15, hi = lane >> 4;
  const int bh = b * HEADS + h;
  const half_t* Qp = qh + (size_t)bh * S_LEN * DK;
  const half_t* Kp = kh + (size_t)bh * S_LEN * DK;
  const half_t* Vp = vh + (size_t)bh * S_LEN * DK;
  const float*  Mp = mask + (size_t)b * S_LEN * S_LEN;

  v8h vreg[2];
#if !ASYNC_LDS
  v8h kreg[2]; v4f mreg[4];
#endif

  auto issue = [&](int t0, int buf) {
#pragma unroll
    for (int ii = 0; ii < 2; ++ii) {
      const int i = tid + ii * 128;
      const int row = i >> 3, c8 = (i & 7) * 8;
      vreg[ii] = *(const v8h*)(Vp + (size_t)(t0 + row) * DK + c8);    // V: register path (transposed)
#if ASYNC_LDS
      __builtin_amdgcn_global_load_async_to_lds_b128(
          AS1I4(Kp + (size_t)(t0 + row) * DK + c8),
          AS3I4(&sK[buf][row * 72 + c8]), 0, 0);
#else
      kreg[ii] = *(const v8h*)(Kp + (size_t)(t0 + row) * DK + c8);
#endif
    }
#pragma unroll
    for (int ii = 0; ii < 4; ++ii) {
      const int i = tid + ii * 128;
      const int row = i >> 3, c4 = (i & 7) * 4;
#if ASYNC_LDS
      __builtin_amdgcn_global_load_async_to_lds_b128(
          AS1I4(Mp + (size_t)(s0 + row) * S_LEN + t0 + c4),
          AS3I4(&sM[buf][row * 36 + c4]), 0, 0);
#else
      mreg[ii] = *(const v4f*)(Mp + (size_t)(s0 + row) * S_LEN + t0 + c4);
#endif
    }
  };
  auto commit = [&](int buf) {
#pragma unroll
    for (int ii = 0; ii < 2; ++ii) {
      const int i = tid + ii * 128;
      const int row = i >> 3, c8 = (i & 7) * 8;
#pragma unroll
      for (int j = 0; j < 8; ++j) sVt[buf][(c8 + j) * 40 + row] = vreg[ii][j];
#if !ASYNC_LDS
      *(v8h*)(&sK[buf][row * 72 + c8]) = kreg[ii];
#endif
    }
#if !ASYNC_LDS
#pragma unroll
    for (int ii = 0; ii < 4; ++ii) {
      const int i = tid + ii * 128;
      const int row = i >> 3, c4 = (i & 7) * 4;
      *(v4f*)(&sM[buf][row * 36 + c4]) = mreg[ii];
    }
#endif
  };

  // stage Q tile once + chunk 0
#pragma unroll
  for (int ii = 0; ii < 4; ++ii) {
    const int i = tid + ii * 128;
    const int row = i >> 3, c8 = (i & 7) * 8;
    v8h q8 = *(const v8h*)(Qp + (size_t)(s0 + row) * DK + c8);
    *(v8h*)(sQ + row * 72 + c8) = q8;
  }
  issue(0, 0);
  commit(0);
#if ASYNC_LDS
  __builtin_amdgcn_s_wait_asynccnt(0);
#endif
  __syncthreads();
  v16h qf0 = load_frag_a(sQ, w * 16, 72, 0);
  v16h qf1 = load_frag_a(sQ, w * 16, 72, 32);

  v8f o[4] = {};
  float rm[8], rl[8];
#pragma unroll
  for (int r = 0; r < 8; ++r) { rm[r] = -1e30f; rl[r] = 0.0f; }

  for (int tc = 0; tc < NCHUNK; ++tc) {
    const int cur = tc & 1;
    const bool more = (tc + 1 < NCHUNK);
    if (more) issue((tc + 1) * 32, cur ^ 1);          // prefetch next chunk

    // scores = (Q K^T)/sqrt(dk) * m - EPS*(1-m), two 16-col tiles from sK[cur]
    v8f sc[2];
#pragma unroll
    for (int st = 0; st < 2; ++st) {
      v16h b0 = load_frag_b(sK[cur], st * 16, 72, 0);
      v16h b1 = load_frag_b(sK[cur], st * 16, 72, 32);
      v8f a = {};
      a = wmma_f16(qf0, b0, a);
      a = wmma_f16(qf1, b1, a);
      const int tcol = st * 16 + nl;
#pragma unroll
      for (int r = 0; r < 8; ++r) {
        float mv = sM[cur][(w * 16 + r + 8 * hi) * 36 + tcol];
        float s  = a[r] * 0.125f;                     // 1/sqrt(64)
        a[r] = s * mv - EPS_MASK * (1.0f - mv);
      }
      sc[st] = a;
    }

    // online softmax (row = acc element r + 8*hi; each row lives in one 16-lane half)
    float cm[8], al[8], cs[8];
#pragma unroll
    for (int r = 0; r < 8; ++r) cm[r] = fmaxf(sc[0][r], sc[1][r]);
#pragma unroll
    for (int off = 1; off < 16; off <<= 1)
#pragma unroll
      for (int r = 0; r < 8; ++r) cm[r] = fmaxf(cm[r], __shfl_xor(cm[r], off, 32));
#pragma unroll
    for (int r = 0; r < 8; ++r) {
      float mn = fmaxf(rm[r], cm[r]);
      al[r] = __expf(rm[r] - mn);
      rm[r] = mn;
    }
#pragma unroll
    for (int st = 0; st < 2; ++st)
#pragma unroll
      for (int r = 0; r < 8; ++r) sc[st][r] = __expf(sc[st][r] - rm[r]);
#pragma unroll
    for (int r = 0; r < 8; ++r) cs[r] = sc[0][r] + sc[1][r];
#pragma unroll
    for (int off = 1; off < 16; off <<= 1)
#pragma unroll
      for (int r = 0; r < 8; ++r) cs[r] += __shfl_xor(cs[r], off, 32);
#pragma unroll
    for (int r = 0; r < 8; ++r) rl[r] = rl[r] * al[r] + cs[r];
#pragma unroll
    for (int vt = 0; vt < 4; ++vt)
#pragma unroll
      for (int r = 0; r < 8; ++r) o[vt][r] *= al[r];

    // P: C-layout -> LDS (wave-private)
#pragma unroll
    for (int st = 0; st < 2; ++st)
#pragma unroll
      for (int r = 0; r < 8; ++r)
        sP[w][(r + 8 * hi) * 40 + st * 16 + nl] = (half_t)sc[st][r];

    if (more) commit(cur ^ 1);                        // V transpose + (sync path) K/mask stores

    // O += P @ V from sVt[cur]
    v16h pa = load_frag_a(sP[w], 0, 40, 0);
#pragma unroll
    for (int vt = 0; vt < 4; ++vt) {
      v16h vb = load_frag_b(sVt[cur], vt * 16, 40, 0);
      o[vt] = wmma_f16(pa, vb, o[vt]);
    }

#if ASYNC_LDS
    __builtin_amdgcn_s_wait_asynccnt(0);
#endif
    __syncthreads();                                  // single barrier per iteration
  }

  // normalize and write ctx[b][s][h*64+dk] (f16)
  float inv[8];
#pragma unroll
  for (int r = 0; r < 8; ++r) inv[r] = 1.0f / rl[r];
#pragma unroll
  for (int vt = 0; vt < 4; ++vt) {
    const int col = h * DK + vt * 16 + nl;
#pragma unroll
    for (int r = 0; r < 8; ++r) {
      const int s = s0 + w * 16 + r + 8 * hi;
      ctx[((size_t)b * S_LEN + s) * DMODEL + col] = (half_t)(o[vt][r] * inv[r]);
    }
  }
}

// ---------------- Kernel 3: output projection, double-buffered, f32 out + bias ----------------
__global__ void __launch_bounds__(128)
out_proj_kernel(const half_t* __restrict__ ctxh, const float* __restrict__ Wo,
                const float* __restrict__ bo, float* __restrict__ out)
{
  __shared__ half_t sA[2][64 * 40];
  __shared__ half_t sBt[2][64 * 40];

  const int n0 = blockIdx.y * 64;
  const int m0 = blockIdx.x * 64;
  const int tid = threadIdx.x, lane = tid & 31, w = tid >> 5;
  const int nl = lane & 15, hi = lane >> 4;

  v8h ar[2]; v4f wr[4];
  auto issue = [&](int kb) {
#pragma unroll
    for (int ii = 0; ii < 2; ++ii) {
      const int i = tid + ii * 128;
      const int row = i >> 2, c8 = (i & 3) * 8;
      ar[ii] = *(const v8h*)(ctxh + (size_t)(m0 + row) * DMODEL + kb + c8);
    }
#pragma unroll
    for (int ii = 0; ii < 4; ++ii) {
      const int i = tid + ii * 128;
      const int d = i >> 4, n4 = (i & 15) * 4;
      wr[ii] = *(const v4f*)(Wo + (size_t)(kb + d) * DMODEL + n0 + n4);
    }
  };
  auto commit = [&](int buf) {
#pragma unroll
    for (int ii = 0; ii < 2; ++ii) {
      const int i = tid + ii * 128;
      const int row = i >> 2, c8 = (i & 3) * 8;
      *(v8h*)(&sA[buf][row * 40 + c8]) = ar[ii];
    }
#pragma unroll
    for (int ii = 0; ii < 4; ++ii) {
      const int i = tid + ii * 128;
      const int d = i >> 4, n4 = (i & 15) * 4;
      sBt[buf][(n4 + 0) * 40 + d] = (half_t)wr[ii].x;
      sBt[buf][(n4 + 1) * 40 + d] = (half_t)wr[ii].y;
      sBt[buf][(n4 + 2) * 40 + d] = (half_t)wr[ii].z;
      sBt[buf][(n4 + 3) * 40 + d] = (half_t)wr[ii].w;
    }
  };

  v8f acc[4] = {};
  issue(0); commit(0);
  __syncthreads();
  for (int kt = 0; kt < NKB; ++kt) {
    const int cur = kt & 1;
    if (kt + 1 < NKB) issue((kt + 1) * 32);
    v16h a = load_frag_a(sA[cur], w * 16, 40, 0);
#pragma unroll
    for (int vt = 0; vt < 4; ++vt) {
      v16h bf = load_frag_b(sBt[cur], vt * 16, 40, 0);
      acc[vt] = wmma_f16(a, bf, acc[vt]);
    }
    if (kt + 1 < NKB) commit(cur ^ 1);
    __syncthreads();
  }
#pragma unroll
  for (int vt = 0; vt < 4; ++vt) {
    const int n = n0 + vt * 16 + nl;
    const float bb_ = bo[n];
#pragma unroll
    for (int r = 0; r < 8; ++r) {
      const int gm = m0 + w * 16 + r + 8 * hi;
      out[(size_t)gm * DMODEL + n] = acc[vt][r] + bb_;
    }
  }
}

// ---------------- Host entry ----------------
extern "C" void kernel_launch(void* const* d_in, const int* in_sizes, int n_in,
                              void* d_out, int out_size, void* d_ws, size_t ws_size,
                              hipStream_t stream) {
  (void)in_sizes; (void)n_in; (void)out_size; (void)ws_size;
  const float* query = (const float*)d_in[0];
  const float* key_  = (const float*)d_in[1];
  const float* value = (const float*)d_in[2];
  const float* mask  = (const float*)d_in[3];
  const float* Wq    = (const float*)d_in[4];
  const float* bq    = (const float*)d_in[5];
  const float* Wk    = (const float*)d_in[6];
  const float* bk    = (const float*)d_in[7];
  const float* Wv    = (const float*)d_in[8];
  const float* bv    = (const float*)d_in[9];
  const float* Wo    = (const float*)d_in[10];
  const float* bo    = (const float*)d_in[11];
  float* out = (float*)d_out;

  const size_t per = (size_t)BATCH * HEADS * S_LEN * DK;   // 2,097,152 f16 each
  half_t* qh  = (half_t*)d_ws;
  half_t* kh  = qh + per;
  half_t* vh  = kh + per;
  half_t* ctx = vh + per;                                  // 16.8 MB of d_ws

  qkv_proj_kernel<<<dim3((BATCH * S_LEN) / 64, HEADS, 3), 128, 0, stream>>>(
      query, key_, value, Wq, bq, Wk, bk, Wv, bv, qh, kh, vh);
  attention_kernel<<<dim3(S_LEN / 64, HEADS, BATCH), 128, 0, stream>>>(
      qh, kh, vh, mask, ctx);
  out_proj_kernel<<<dim3((BATCH * S_LEN) / 64, DMODEL / 64, 1), 128, 0, stream>>>(
      ctx, Wo, bo, out);
}
